// ExphormerFullLayer_11476152615032
// MI455X (gfx1250) — matrix operand order repaired and checked by reading, not compiled, not run on hardware
//
#include <hip/hip_runtime.h>
#include <hip/hip_bf16.h>
#include <math.h>

// ---------------------------------------------------------------------------
// Exphormer layer for MI455X (gfx1250, wave32, WMMA).
// All GEMMs: v_wmma_f32_16x16x32_bf16 (bf16 A/B converted on the fly, f32 acc).
// Edge GEMM fused with attention score + exp + scatter-atomics (Ee never hits HBM).
// ---------------------------------------------------------------------------

#define NN 100000
#define EE 800000
#define DD 64
#define HH 4
#define DHH 16

typedef __attribute__((ext_vector_type(16))) __bf16 v16bf;
typedef __attribute__((ext_vector_type(8)))  float  v8f;
typedef __attribute__((ext_vector_type(4)))  float  f4;

__device__ __forceinline__ v8f wmma_bf16(v16bf a, v16bf b, v8f c) {
  // (neg_a, A, neg_b, B, c_mod, C, reuse_a, reuse_b)
  return __builtin_amdgcn_wmma_f32_16x16x32_bf16(false, a, false, b, (short)0, c, false, false);
}

// A fragment: 16 rows x 32 K, from row-major f32 (converted to bf16).
// lane<16: row=lane,    K = {koff..koff+7, 16+koff..16+koff+7}, koff=0
// lane>=16: row=lane-16, koff=8  (per ISA 16-bit A table)
__device__ __forceinline__ v16bf load_a_frag(const float* __restrict__ src, int ld,
                                             int row0, int kb, int lane) {
  int r    = row0 + (lane & 15);
  int koff = (lane & 16) ? 8 : 0;
  const float* p = src + r * ld + kb + koff;
  float t[16];
  *(f4*)&t[0]  = *(const f4*)(p);
  *(f4*)&t[4]  = *(const f4*)(p + 4);
  *(f4*)&t[8]  = *(const f4*)(p + 16);
  *(f4*)&t[12] = *(const f4*)(p + 20);
  v16bf a;
#pragma unroll
  for (int i = 0; i < 16; ++i) a[i] = (__bf16)t[i];
  return a;
}

// B fragment: 32 K x 16 N from row-major f32 weight [K x ldw].
// lane<16: col=lane, K=kb+{0..7,16..23} ; lane>=16: col=lane-16, K=kb+{8..15,24..31}
__device__ __forceinline__ v16bf load_b_frag(const float* __restrict__ W, int ldw,
                                             int kb, int ncol0, int lane) {
  int n  = ncol0 + (lane & 15);
  int k0 = kb + ((lane & 16) ? 8 : 0);
  v16bf b;
#pragma unroll
  for (int i = 0; i < 16; ++i) b[i] = (__bf16)W[(k0 + (i & 7) + ((i & 8) << 1)) * ldw + n];
  return b;
}

// C tile store: VGPR j holds M=j (lanes<16) / M=8+j (lanes>=16), N=lane&15
__device__ __forceinline__ void store_c(float* __restrict__ dst, int ld,
                                        int row0, int ncol0, int lane, v8f c) {
  int n  = ncol0 + (lane & 15);
  int r0 = row0 + ((lane & 16) ? 8 : 0);
#pragma unroll
  for (int j = 0; j < 8; ++j) dst[(r0 + j) * ld + n] = c[j];
}

// -------------------- zero fill --------------------
__global__ void k_zero(float* p, long n) {
  long i = (long)blockIdx.x * blockDim.x + threadIdx.x;
  long s = (long)gridDim.x * blockDim.x;
  for (; i < n; i += s) p[i] = 0.0f;
}

// -------------------- node projections: Qh|Kh|Vh = x @ W --------------------
__global__ void k_node_proj(const float* __restrict__ x,
                            const float* __restrict__ Wq, const float* __restrict__ Wk,
                            const float* __restrict__ Wv,
                            float* __restrict__ Qh, float* __restrict__ Kh,
                            float* __restrict__ Vh) {
  int lane = threadIdx.x & 31;
  int gw   = blockIdx.x * (blockDim.x >> 5) + (threadIdx.x >> 5);
  int nw   = gridDim.x * (blockDim.x >> 5);
  const float* W;
  float* out;
  if (blockIdx.y == 0)      { W = Wq; out = Qh; }
  else if (blockIdx.y == 1) { W = Wk; out = Kh; }
  else                      { W = Wv; out = Vh; }

  v16bf B[4][2];
#pragma unroll
  for (int nt = 0; nt < 4; ++nt)
#pragma unroll
    for (int kk = 0; kk < 2; ++kk)
      B[nt][kk] = load_b_frag(W, DD, kk * 32, nt * 16, lane);

  for (int rb = gw; rb < NN / 16; rb += nw) {
    int row0 = rb * 16;
    v16bf A0 = load_a_frag(x, DD, row0, 0, lane);
    v16bf A1 = load_a_frag(x, DD, row0, 32, lane);
#pragma unroll
    for (int nt = 0; nt < 4; ++nt) {
      v8f c = {};
      c = wmma_bf16(A0, B[nt][0], c);
      c = wmma_bf16(A1, B[nt][1], c);
      store_c(out, DD, row0, nt * 16, lane, c);
    }
  }
}

// ------ fused edge kernel: Ee = edge_attr@We (WMMA -> LDS), score, scatter ------
__global__ void k_edge(const float* __restrict__ ea, const int* __restrict__ eidx,
                       const float* __restrict__ We,
                       const float* __restrict__ Qh, const float* __restrict__ Kh,
                       const float* __restrict__ Vh,
                       float* __restrict__ wV, float* __restrict__ Zb) {
  __shared__ float lds[4][16 * DD];
  int lane = threadIdx.x & 31;
  int w    = threadIdx.x >> 5;
  int gw   = blockIdx.x * 4 + w;
  int nw   = gridDim.x * 4;
  float* tile = lds[w];

  v16bf B[4][2];
#pragma unroll
  for (int nt = 0; nt < 4; ++nt)
#pragma unroll
    for (int kk = 0; kk < 2; ++kk)
      B[nt][kk] = load_b_frag(We, DD, kk * 32, nt * 16, lane);

  int n     = lane & 15;
  int rbase = (lane & 16) ? 8 : 0;

  for (int eb = gw; eb < EE / 16; eb += nw) {
    int row0 = eb * 16;
    v16bf A0 = load_a_frag(ea, DD, row0, 0, lane);
    v16bf A1 = load_a_frag(ea, DD, row0, 32, lane);
#pragma unroll
    for (int nt = 0; nt < 4; ++nt) {
      v8f c = {};
      c = wmma_bf16(A0, B[nt][0], c);
      c = wmma_bf16(A1, B[nt][1], c);
#pragma unroll
      for (int j = 0; j < 8; ++j) tile[(rbase + j) * DD + nt * 16 + n] = c[j];
    }
    // wave-internal LDS visibility: DS ops are in-order per wave; make the
    // compiler keep order and drain DScnt before cross-lane reads.
    asm volatile("s_wait_dscnt 0" ::: "memory");

#pragma unroll
    for (int it = 0; it < 2; ++it) {
      int p  = it * 32 + lane;   // 16 edges x 4 heads = 64 slots
      int le = p >> 2;
      int h  = p & 3;
      int e  = row0 + le;
      int s  = eidx[e];
      int d  = eidx[EE + e];
      const float* ep = tile + le * DD + h * DHH;
      const float* kp = Kh + s * DD + h * DHH;
      const float* qp = Qh + d * DD + h * DHH;
      float dot = 0.0f;
#pragma unroll
      for (int v = 0; v < 4; ++v) {
        f4 e4 = *(const f4*)(ep + v * 4);
        f4 k4 = *(const f4*)(kp + v * 4);
        f4 q4 = *(const f4*)(qp + v * 4);
        dot += e4.x * k4.x * q4.x + e4.y * k4.y * q4.y +
               e4.z * k4.z * q4.z + e4.w * k4.w * q4.w;
      }
      // score = exp(clip(sum(K*Q/sqrt(DH)*E), -5, 5)), sqrt(16)=4
      float sc = __expf(fminf(fmaxf(dot * 0.25f, -5.0f), 5.0f));
      const float* vp = Vh + s * DD + h * DHH;
      float* wp = wV + d * DD + h * DHH;
#pragma unroll
      for (int v = 0; v < 4; ++v) {
        f4 v4 = *(const f4*)(vp + v * 4);
        unsafeAtomicAdd(wp + v * 4 + 0, v4.x * sc);
        unsafeAtomicAdd(wp + v * 4 + 1, v4.y * sc);
        unsafeAtomicAdd(wp + v * 4 + 2, v4.z * sc);
        unsafeAtomicAdd(wp + v * 4 + 3, v4.w * sc);
      }
      unsafeAtomicAdd(Zb + d * HH + h, sc);
    }
    asm volatile("s_wait_dscnt 0" ::: "memory");
  }
}

// ------- h1 = x + wV/(Z+1e-6) ; accumulate BN1 column sums/sumsq -------
__global__ void k_combine_stats(const float* __restrict__ x, const float* __restrict__ wV,
                                const float* __restrict__ Zb, float* __restrict__ h1,
                                float* __restrict__ sum, float* __restrict__ sumsq) {
  int t  = threadIdx.x;     // 256
  int cg = t & 15;          // 4-column group; head = cg>>2
  int r0 = blockIdx.x * 16 + (t >> 4);
  int rs = gridDim.x * 16;
  f4 s = {}, sq = {};
  for (int r = r0; r < NN; r += rs) {
    f4 xv = *(const f4*)(x  + r * DD + cg * 4);
    f4 wv = *(const f4*)(wV + r * DD + cg * 4);
    float zinv = 1.0f / (Zb[r * HH + (cg >> 2)] + 1e-6f);
    f4 h;
    h.x = xv.x + wv.x * zinv; h.y = xv.y + wv.y * zinv;
    h.z = xv.z + wv.z * zinv; h.w = xv.w + wv.w * zinv;
    *(f4*)(h1 + r * DD + cg * 4) = h;
    s += h; sq += h * h;
  }
  unsafeAtomicAdd(sum + cg * 4 + 0, s.x);  unsafeAtomicAdd(sum + cg * 4 + 1, s.y);
  unsafeAtomicAdd(sum + cg * 4 + 2, s.z);  unsafeAtomicAdd(sum + cg * 4 + 3, s.w);
  unsafeAtomicAdd(sumsq + cg * 4 + 0, sq.x); unsafeAtomicAdd(sumsq + cg * 4 + 1, sq.y);
  unsafeAtomicAdd(sumsq + cg * 4 + 2, sq.z); unsafeAtomicAdd(sumsq + cg * 4 + 3, sq.w);
}

// ------- column stats only (BN2) -------
__global__ void k_stats(const float* __restrict__ h, float* __restrict__ sum,
                        float* __restrict__ sumsq) {
  int t  = threadIdx.x;
  int cg = t & 15;
  int r0 = blockIdx.x * 16 + (t >> 4);
  int rs = gridDim.x * 16;
  f4 s = {}, sq = {};
  for (int r = r0; r < NN; r += rs) {
    f4 hv = *(const f4*)(h + r * DD + cg * 4);
    s += hv; sq += hv * hv;
  }
  unsafeAtomicAdd(sum + cg * 4 + 0, s.x);  unsafeAtomicAdd(sum + cg * 4 + 1, s.y);
  unsafeAtomicAdd(sum + cg * 4 + 2, s.z);  unsafeAtomicAdd(sum + cg * 4 + 3, s.w);
  unsafeAtomicAdd(sumsq + cg * 4 + 0, sq.x); unsafeAtomicAdd(sumsq + cg * 4 + 1, sq.y);
  unsafeAtomicAdd(sumsq + cg * 4 + 2, sq.z); unsafeAtomicAdd(sumsq + cg * 4 + 3, sq.w);
}

// ------- finalize BN: a = g*rsqrt(var+eps), b = beta - mu*a -------
__global__ void k_bn_final(const float* __restrict__ sum, const float* __restrict__ sumsq,
                           const float* __restrict__ g, const float* __restrict__ be,
                           float* __restrict__ a, float* __restrict__ b) {
  int c = threadIdx.x;  // 64
  float mu  = sum[c] * (1.0f / NN);
  float var = sumsq[c] * (1.0f / NN) - mu * mu;
  float inv = rsqrtf(var + 1e-5f);
  float ac  = g[c] * inv;
  a[c] = ac;
  b[c] = be[c] - mu * ac;
}

// ------- per-column affine apply: dst = src*a[col] + b[col] -------
__global__ void k_affine(const float* __restrict__ src, const float* __restrict__ a,
                         const float* __restrict__ b, float* __restrict__ dst, long count) {
  long g = (long)blockIdx.x * blockDim.x + threadIdx.x;  // f4 groups
  long s = (long)gridDim.x * blockDim.x;
  long ng = count >> 2;
  for (; g < ng; g += s) {
    int col0 = (int)((g << 2) & (DD - 1));
    f4 av = *(const f4*)(a + col0);
    f4 bv = *(const f4*)(b + col0);
    f4 x  = *(const f4*)(src + (g << 2));
    f4 y;
    y.x = x.x * av.x + bv.x; y.y = x.y * av.y + bv.y;
    y.z = x.z * av.z + bv.z; y.w = x.w * av.w + bv.w;
    *(f4*)(dst + (g << 2)) = y;
  }
}

// ------- FFN layer 1: hidden = relu(hn @ W1 + b1), W1: 64x128 -------
__global__ void k_ffn1(const float* __restrict__ hn, const float* __restrict__ W1,
                       const float* __restrict__ b1, float* __restrict__ hidden) {
  int lane  = threadIdx.x & 31;
  int gw    = blockIdx.x * (blockDim.x >> 5) + (threadIdx.x >> 5);
  int nw    = gridDim.x * (blockDim.x >> 5);
  int nbase = blockIdx.y * 64;  // which 64-col half of the 128

  v16bf B[4][2];
#pragma unroll
  for (int nt = 0; nt < 4; ++nt)
#pragma unroll
    for (int kk = 0; kk < 2; ++kk)
      B[nt][kk] = load_b_frag(W1, 2 * DD, kk * 32, nbase + nt * 16, lane);

  float bias[4];
#pragma unroll
  for (int nt = 0; nt < 4; ++nt) bias[nt] = b1[nbase + nt * 16 + (lane & 15)];

  for (int rb = gw; rb < NN / 16; rb += nw) {
    int row0 = rb * 16;
    v16bf A0 = load_a_frag(hn, DD, row0, 0, lane);
    v16bf A1 = load_a_frag(hn, DD, row0, 32, lane);
#pragma unroll
    for (int nt = 0; nt < 4; ++nt) {
      v8f c = {};
      c = wmma_bf16(A0, B[nt][0], c);
      c = wmma_bf16(A1, B[nt][1], c);
      int nn = nbase + nt * 16 + (lane & 15);
      int r0 = row0 + ((lane & 16) ? 8 : 0);
#pragma unroll
      for (int j = 0; j < 8; ++j)
        hidden[(r0 + j) * (2 * DD) + nn] = fmaxf(c[j] + bias[nt], 0.0f);
    }
  }
}

// ------- FFN layer 2 + residual: h2 = hn + hidden@W2 + b2, W2: 128x64 -------
__global__ void k_ffn2(const float* __restrict__ hidden, const float* __restrict__ W2,
                       const float* __restrict__ b2, const float* __restrict__ hn,
                       float* __restrict__ h2) {
  int lane  = threadIdx.x & 31;
  int gw    = blockIdx.x * (blockDim.x >> 5) + (threadIdx.x >> 5);
  int nw    = gridDim.x * (blockDim.x >> 5);
  int nbase = blockIdx.y * 32;  // 2 tiles of 16 cols each

  v16bf B[2][4];
#pragma unroll
  for (int nt = 0; nt < 2; ++nt)
#pragma unroll
    for (int kk = 0; kk < 4; ++kk)
      B[nt][kk] = load_b_frag(W2, DD, kk * 32, nbase + nt * 16, lane);

  float bias[2];
#pragma unroll
  for (int nt = 0; nt < 2; ++nt) bias[nt] = b2[nbase + nt * 16 + (lane & 15)];

  for (int rb = gw; rb < NN / 16; rb += nw) {
    int row0 = rb * 16;
    v16bf A[4];
#pragma unroll
    for (int kk = 0; kk < 4; ++kk)
      A[kk] = load_a_frag(hidden, 2 * DD, row0, kk * 32, lane);
#pragma unroll
    for (int nt = 0; nt < 2; ++nt) {
      v8f c = {};
#pragma unroll
      for (int kk = 0; kk < 4; ++kk) c = wmma_bf16(A[kk], B[nt][kk], c);
      int nn = nbase + nt * 16 + (lane & 15);
      int r0 = row0 + ((lane & 16) ? 8 : 0);
#pragma unroll
      for (int j = 0; j < 8; ++j) {
        float v = c[j] + bias[nt] + hn[(r0 + j) * DD + nn];
        h2[(r0 + j) * DD + nn] = v;
      }
    }
  }
}

// ---------------------------------------------------------------------------
extern "C" void kernel_launch(void* const* d_in, const int* in_sizes, int n_in,
                              void* d_out, int out_size, void* d_ws, size_t ws_size,
                              hipStream_t stream) {
  const float* x   = (const float*)d_in[0];
  const float* ea  = (const float*)d_in[1];
  const int*   eix = (const int*)d_in[2];
  const float* Wq  = (const float*)d_in[3];
  const float* Wk  = (const float*)d_in[4];
  const float* We  = (const float*)d_in[5];
  const float* Wv  = (const float*)d_in[6];
  const float* W1  = (const float*)d_in[7];
  const float* b1  = (const float*)d_in[8];
  const float* W2  = (const float*)d_in[9];
  const float* b2  = (const float*)d_in[10];
  const float* g1  = (const float*)d_in[11];
  const float* be1 = (const float*)d_in[12];
  const float* g2  = (const float*)d_in[13];
  const float* be2 = (const float*)d_in[14];
  float* out = (float*)d_out;

  const long NQ = (long)NN * DD;
  float* w      = (float*)d_ws;
  float* Qh     = w;
  float* Kh     = Qh + NQ;
  float* Vh     = Kh + NQ;
  float* wV     = Vh + NQ;
  float* Zb     = wV + NQ;
  float* stats  = Zb + (long)NN * HH;  // sum1|sq1|sum2|sq2 (4*64)
  float* coef   = stats + 256;         // a1|b1s|a2|b2s (4*64)
  float* h1     = coef + 256;
  float* hn     = h1 + NQ;
  float* hidden = hn + NQ;
  float* h2     = hidden + (long)NN * (2 * DD);

  long zcount = NQ + (long)NN * HH + 256;  // wV, Zb, stats

  k_zero<<<dim3(1024), 256, 0, stream>>>(wV, zcount);
  k_node_proj<<<dim3(256, 3), 128, 0, stream>>>(x, Wq, Wk, Wv, Qh, Kh, Vh);
  k_edge<<<dim3(1024), 128, 0, stream>>>(ea, eix, We, Qh, Kh, Vh, wV, Zb);
  k_combine_stats<<<dim3(512), 256, 0, stream>>>(x, wV, Zb, h1, stats, stats + 64);
  k_bn_final<<<dim3(1), 64, 0, stream>>>(stats, stats + 64, g1, be1, coef, coef + 64);
  k_affine<<<dim3(512), 256, 0, stream>>>(h1, coef, coef + 64, hn, NQ);
  k_ffn1<<<dim3(256, 2), 128, 0, stream>>>(hn, W1, b1, hidden);
  k_ffn2<<<dim3(256, 2), 128, 0, stream>>>(hidden, W2, b2, hn, h2);
  k_stats<<<dim3(512), 256, 0, stream>>>(h2, stats + 128, stats + 192);
  k_bn_final<<<dim3(1), 64, 0, stream>>>(stats + 128, stats + 192, g2, be2,
                                         coef + 128, coef + 192);
  k_affine<<<dim3(512), 256, 0, stream>>>(h2, coef + 128, coef + 192, out, NQ);
}
